// DilateAttention_54898271977738
// MI455X (gfx1250) — matrix operand order repaired and checked by reading, not compiled, not run on hardware
//
#include <hip/hip_runtime.h>
#include <hip/hip_bf16.h>

// ---------------------------------------------------------------------------
// Dilated multi-head attention for MI455X (gfx1250), bf16 WMMA throughout.
//   B=2, S=4096, E=1024, H=16, D=64, DILATION=4 -> Sk=1024 keys/head.
// Dilation is folded into the K/V projections (x[::4] @ W), saving 3/4 of
// those GEMMs. All matmuls use v_wmma_f32_16x16x32_bf16. Tile staging uses
// the gfx1250 async global->LDS DMA path (ASYNCcnt) with double buffering,
// and fragment loads are batched ahead of each WMMA chain so LDS latency is
// paid once per tile, not once per WMMA.
// ---------------------------------------------------------------------------

typedef __bf16 bf16;
typedef __attribute__((ext_vector_type(16))) __bf16 v16bf;
typedef __attribute__((ext_vector_type(8)))  float  v8f;

#define E_DIM   1024
#define S_LEN   4096
#define B_SZ    2
#define H_NUM   16
#define D_HEAD  64
#define DIL     4
#define SK_LEN  (S_LEN / DIL)

union FragU { v16bf v; uint4 q[2]; };

// ---- gfx1250 async global->LDS copy (16B per lane, tracked by ASYNCcnt) ----
// VGPR operand 0 = LDS byte offset (flat shared address truncates to the LDS
// offset, ISA 10.2), operand 1 = 64-bit global address.
__device__ __forceinline__ void async_copy_b128(unsigned lds_off, const bf16* gptr) {
    asm volatile("global_load_async_to_lds_b128 %0, %1, off"
                 :: "v"(lds_off), "v"(gptr)
                 : "memory");
}

__device__ __forceinline__ void wait_async0() {
#if __has_builtin(__builtin_amdgcn_s_wait_asynccnt)
    __builtin_amdgcn_s_wait_asynccnt(0);
#else
    asm volatile("s_wait_asynccnt 0x0" ::: "memory");
#endif
}

// A-matrix fragment (16x32 bf16, ISA 7.12.2): lane l -> row (l&15);
// byte chunks at (l>>4)*16 (K 0..7 / 8..15) and +32 (K 16..23 / 24..31).
__device__ __forceinline__ v16bf load_a_frag(const bf16* base, int stride_elems) {
    const int l = threadIdx.x & 31;
    const char* p = (const char*)(base + (size_t)(l & 15) * stride_elems) + ((l >> 4) << 4);
    FragU u;
    u.q[0] = *(const uint4*)(p);
    u.q[1] = *(const uint4*)(p + 32);
    return u.v;
}

// B-matrix fragment (32x16 bf16): lane l -> column (l&15); byte chunks at
// (l>>4)*16 (K 0..7 / 8..15) and +32 (K 16..23 / 24..31), matching A.
// `base` points at [n0][k0] of an N-major (transposed) tile.
__device__ __forceinline__ v16bf load_b_frag(const bf16* base, int stride_elems) {
    const int l = threadIdx.x & 31;
    const char* p = (const char*)(base + (size_t)(l & 15) * stride_elems) + ((l >> 4) << 4);
    FragU u;
    u.q[0] = *(const uint4*)(p);
    u.q[1] = *(const uint4*)(p + 32);
    return u.v;
}

__device__ __forceinline__ v8f wmma_bf16(v16bf a, v16bf b, v8f c) {
    return __builtin_amdgcn_wmma_f32_16x16x32_bf16(false, a, false, b, (short)0, c,
                                                   false, false);
}

__device__ __forceinline__ v8f v8f_zero() {
    v8f z;
#pragma unroll
    for (int i = 0; i < 8; ++i) z[i] = 0.0f;
    return z;
}

// ---------------------------------------------------------------------------
// fp32 -> bf16 bulk convert (vectorized by 4)
// ---------------------------------------------------------------------------
__global__ void __launch_bounds__(256)
cvt_f32_to_bf16(const float* __restrict__ in, bf16* __restrict__ out, int n4) {
    int i = blockIdx.x * blockDim.x + threadIdx.x;
    if (i >= n4) return;
    float4 f = ((const float4*)in)[i];
    int b = i * 4;
    out[b + 0] = (bf16)f.x;
    out[b + 1] = (bf16)f.y;
    out[b + 2] = (bf16)f.z;
    out[b + 3] = (bf16)f.w;
}

// ---------------------------------------------------------------------------
// Transpose E x E fp32 weight into N-major bf16 (Wt[n*E + k] = W[k*E + n])
// ---------------------------------------------------------------------------
__global__ void __launch_bounds__(256)
transpose_to_bf16(const float* __restrict__ W, bf16* __restrict__ Wt, int E) {
    __shared__ float tile[32][33];
    const int n0 = blockIdx.x * 32;
    const int k0 = blockIdx.y * 32;
    const int tx = threadIdx.x;   // 0..31
    const int ty = threadIdx.y;   // 0..7
#pragma unroll
    for (int i = 0; i < 32; i += 8)
        tile[ty + i][tx] = W[(size_t)(k0 + ty + i) * E + n0 + tx];
    __syncthreads();
#pragma unroll
    for (int i = 0; i < 32; i += 8)
        Wt[(size_t)(n0 + ty + i) * E + k0 + tx] = (bf16)tile[tx][ty + i];
}

// ---------------------------------------------------------------------------
// Tiled GEMM:  C[M x N] = rowmap(A)[M x K] * Bt^T + bias
//   A  : bf16; logical row r maps to physical row
//        (r / rows_per_b) * (rows_per_b * dil) + (r % rows_per_b) * dil
//   Bt : bf16, N x K (N-major)
// Block: 256 threads = 8 waves; tile 128x128; K-step 32; double-buffered LDS
// filled by async global->LDS DMA overlapped with WMMA on the other buffer.
// ---------------------------------------------------------------------------
#define GE_TM 128
#define GE_TN 128
#define GE_TK 32
#define LDS_STR 48   // 32 + 16 pad elems -> 96B row pitch (16B aligned)

__global__ void __launch_bounds__(256)
gemm_bf16_wmma(const bf16* __restrict__ A, const bf16* __restrict__ Bt,
               const float* __restrict__ bias,
               float* __restrict__ Cf, bf16* __restrict__ Cb,
               int M, int N, int K, int rows_per_b, int dil) {
    __shared__ bf16 As[2][GE_TM * LDS_STR];
    __shared__ bf16 Bs[2][GE_TN * LDS_STR];

    const int tid  = threadIdx.x;
    const int wave = tid >> 5;
    const int l    = tid & 31;
    const int hi   = l >> 4;
    const int nl   = l & 15;
    const int m0   = blockIdx.x * GE_TM;
    const int n0   = blockIdx.y * GE_TN;

    // Per-thread staging coordinates: 2 x 16B chunks each for A and B tiles
    // (128 rows x 32 K-elems = 512 chunks, 256 threads).
    int rowc[2], ccc[2];
    const bf16* aG[2];
    const bf16* bG[2];
    unsigned ldsA[2][2], ldsB[2][2];
#pragma unroll
    for (int c = 0; c < 2; ++c) {
        const int chunk = tid + c * 256;
        rowc[c] = chunk >> 2;
        ccc[c]  = chunk & 3;
        const int r  = m0 + rowc[c];
        const int ar = (r / rows_per_b) * (rows_per_b * dil) + (r % rows_per_b) * dil;
        aG[c] = A  + (size_t)ar * K + ccc[c] * 8;
        bG[c] = Bt + (size_t)(n0 + rowc[c]) * K + ccc[c] * 8;
#pragma unroll
        for (int bb = 0; bb < 2; ++bb) {
            ldsA[bb][c] = (unsigned)(uintptr_t)&As[bb][rowc[c] * LDS_STR + ccc[c] * 8];
            ldsB[bb][c] = (unsigned)(uintptr_t)&Bs[bb][rowc[c] * LDS_STR + ccc[c] * 8];
        }
    }

    v8f acc[8];
#pragma unroll
    for (int i = 0; i < 8; ++i) acc[i] = v8f_zero();

    // Prologue: DMA tile 0 into buffer 0.
#pragma unroll
    for (int c = 0; c < 2; ++c) {
        async_copy_b128(ldsA[0][c], aG[c]);
        async_copy_b128(ldsB[0][c], bG[c]);
    }

    int cur = 0;
    for (int k0 = 0; k0 < K; k0 += GE_TK) {
        wait_async0();        // our buffer's DMA done (per-wave)
        __syncthreads();      // everyone's DMA done; prior reads of buf^1 done

        if (k0 + GE_TK < K) { // kick off DMA for next K-slab into other buffer
#pragma unroll
            for (int c = 0; c < 2; ++c) {
                async_copy_b128(ldsA[cur ^ 1][c], aG[c] + k0 + GE_TK);
                async_copy_b128(ldsB[cur ^ 1][c], bG[c] + k0 + GE_TK);
            }
        }

        // Batch all fragment loads ahead of the WMMA chain: one dscnt wait
        // per tile instead of one per WMMA.
        v16bf a = load_a_frag(&As[cur][(wave * 16) * LDS_STR], LDS_STR);
        v16bf bfr[8];
#pragma unroll
        for (int nt = 0; nt < 8; ++nt)
            bfr[nt] = load_b_frag(&Bs[cur][(nt * 16) * LDS_STR], LDS_STR);
#pragma unroll
        for (int nt = 0; nt < 8; ++nt)
            acc[nt] = wmma_bf16(a, bfr[nt], acc[nt]);
        cur ^= 1;
    }

    // Epilogue: C-frag layout: VGPR j, lane l -> row j+8*(l>=16), col (l&15).
#pragma unroll
    for (int nt = 0; nt < 8; ++nt) {
        const int col = n0 + nt * 16 + nl;
        const float bv = bias ? bias[col] : 0.0f;
#pragma unroll
        for (int j = 0; j < 8; ++j) {
            const int row = m0 + wave * 16 + j + 8 * hi;
            const float v = acc[nt][j] + bv;
            if (Cf) Cf[(size_t)row * N + col] = v;
            if (Cb) Cb[(size_t)row * N + col] = (bf16)v;
        }
    }
}

// ---------------------------------------------------------------------------
// Flash-style dilated attention.
// Grid: (S/128 query tiles, B*H). Block: 256 threads = 8 waves; wave w owns
// 16 query rows. Loop over Sk=1024 keys in chunks of 64:
//   - K chunk staged row-major via async global->LDS DMA
//   - V chunk staged TRANSPOSED (Vt[d][key]) so P@V B-fragments read
//     contiguous K
//   - scores via 2 WMMA k-steps/tile; online softmax with shfl_xor half-wave
//     reductions; P spilled bf16 to per-wave LDS pane for the P@V WMMAs.
// ---------------------------------------------------------------------------
#define CHUNK   64
#define KS_STR  80   // 64 + 16 pad elems -> 160B pitch (16B aligned)

__global__ void __launch_bounds__(256)
attn_dilated_wmma(const bf16* __restrict__ Q, const bf16* __restrict__ Kd,
                  const bf16* __restrict__ Vd, bf16* __restrict__ Ctx) {
    __shared__ bf16 Ks[CHUNK * KS_STR];            // [key][d]
    __shared__ bf16 Vt[CHUNK * KS_STR];            // [d][key]
    __shared__ bf16 Ps[8 * 16 * KS_STR];           // per-wave 16 x 64 P tile

    const int tid  = threadIdx.x;
    const int wave = tid >> 5;
    const int l    = tid & 31;
    const int hi   = l >> 4;
    const int nl   = l & 15;

    const int qt = blockIdx.x;            // query tile (128 rows)
    const int bh = blockIdx.y;
    const int b  = bh >> 4;
    const int h  = bh & 15;

    const bf16* qbase = Q  + (size_t)b * S_LEN  * E_DIM + h * D_HEAD;
    const bf16* kbase = Kd + (size_t)b * SK_LEN * E_DIM + h * D_HEAD;
    const bf16* vbase = Vd + (size_t)b * SK_LEN * E_DIM + h * D_HEAD;

    const int qrow0 = qt * 128 + wave * 16;

    // Persistent Q fragments: K-windows [0,32) and [32,64) of D=64.
    const v16bf aq0 = load_a_frag(qbase + (size_t)qrow0 * E_DIM + 0,  E_DIM);
    const v16bf aq1 = load_a_frag(qbase + (size_t)qrow0 * E_DIM + 32, E_DIM);

    v8f o[4];
#pragma unroll
    for (int dt = 0; dt < 4; ++dt) o[dt] = v8f_zero();
    float mrun[8], lrun[8];
#pragma unroll
    for (int j = 0; j < 8; ++j) { mrun[j] = -1e30f; lrun[j] = 0.0f; }

    const float scale = 0.125f;  // 1/sqrt(64)

    for (int c0 = 0; c0 < SK_LEN; c0 += CHUNK) {
        __syncthreads();   // previous iteration's reads of Ks/Vt complete
        // Stage K (async DMA) and V (manual transpose): 64 rows x 128B = 512
        // 16B chunks each; 2 per thread.
#pragma unroll
        for (int c = 0; c < 2; ++c) {
            const int chunk = tid + c * 256;
            const int row   = chunk >> 3;      // key 0..63
            const int cc    = chunk & 7;       // 16B chunk in row
            async_copy_b128((unsigned)(uintptr_t)&Ks[row * KS_STR + cc * 8],
                            kbase + (size_t)(c0 + row) * E_DIM + cc * 8);

            union { uint4 q; bf16 e[8]; } vv;
            vv.q = *(const uint4*)(vbase + (size_t)(c0 + row) * E_DIM + cc * 8);
#pragma unroll
            for (int i = 0; i < 8; ++i)
                Vt[(cc * 8 + i) * KS_STR + row] = vv.e[i];
        }
        wait_async0();
        __syncthreads();

        // Scores: preload all 8 K fragments, then the 8 WMMAs.
        v16bf kb[4][2];
#pragma unroll
        for (int nt = 0; nt < 4; ++nt) {
            kb[nt][0] = load_b_frag(&Ks[nt * 16 * KS_STR + 0],  KS_STR);
            kb[nt][1] = load_b_frag(&Ks[nt * 16 * KS_STR + 32], KS_STR);
        }
        v8f s[4];
#pragma unroll
        for (int nt = 0; nt < 4; ++nt) {
            s[nt] = v8f_zero();
            s[nt] = wmma_bf16(aq0, kb[nt][0], s[nt]);
            s[nt] = wmma_bf16(aq1, kb[nt][1], s[nt]);
        }

        // Online softmax; each (j, lane-half) pair is one query row.
#pragma unroll
        for (int j = 0; j < 8; ++j) {
            float cm = fmaxf(fmaxf(s[0][j], s[1][j]), fmaxf(s[2][j], s[3][j]));
#pragma unroll
            for (int mk = 1; mk < 16; mk <<= 1)
                cm = fmaxf(cm, __shfl_xor(cm, mk, 32));
            const float nm = fmaxf(mrun[j], cm);
            const float al = __expf(scale * (mrun[j] - nm));
            float rs = 0.0f;
#pragma unroll
            for (int nt = 0; nt < 4; ++nt) {
                const float p = __expf(scale * (s[nt][j] - nm));
                rs += p;
                Ps[(wave * 16 + j + 8 * hi) * KS_STR + nt * 16 + nl] = (bf16)p;
            }
#pragma unroll
            for (int mk = 1; mk < 16; mk <<= 1)
                rs += __shfl_xor(rs, mk, 32);
            lrun[j] = lrun[j] * al + rs;
            mrun[j] = nm;
#pragma unroll
            for (int dt = 0; dt < 4; ++dt) o[dt][j] *= al;
        }
        __syncthreads();   // P visible before A-fragment reads

        // O += P @ V : P fragments loaded once; all 8 V fragments preloaded.
        const bf16* Pw = &Ps[wave * 16 * KS_STR];
        v16bf ap0 = load_a_frag(Pw + 0,  KS_STR);
        v16bf ap1 = load_a_frag(Pw + 32, KS_STR);
        v16bf vb[4][2];
#pragma unroll
        for (int dt = 0; dt < 4; ++dt) {
            vb[dt][0] = load_b_frag(&Vt[dt * 16 * KS_STR + 0],  KS_STR);
            vb[dt][1] = load_b_frag(&Vt[dt * 16 * KS_STR + 32], KS_STR);
        }
#pragma unroll
        for (int dt = 0; dt < 4; ++dt) {
            o[dt] = wmma_bf16(ap0, vb[dt][0], o[dt]);
            o[dt] = wmma_bf16(ap1, vb[dt][1], o[dt]);
        }
    }

    // Normalize and store ctx (bf16, [B,S,E] with head-h columns).
#pragma unroll
    for (int dt = 0; dt < 4; ++dt) {
#pragma unroll
        for (int j = 0; j < 8; ++j) {
            const int row = qrow0 + j + 8 * hi;
            const float val = o[dt][j] / lrun[j];
            Ctx[((size_t)(b * S_LEN + row)) * E_DIM + h * D_HEAD + dt * 16 + nl] =
                (bf16)val;
        }
    }
}

// ---------------------------------------------------------------------------
// Host-side orchestration
// ---------------------------------------------------------------------------
extern "C" void kernel_launch(void* const* d_in, const int* in_sizes, int n_in,
                              void* d_out, int out_size, void* d_ws, size_t ws_size,
                              hipStream_t stream) {
    const float* x  = (const float*)d_in[0];
    const float* Wq = (const float*)d_in[1];
    const float* bq = (const float*)d_in[2];
    const float* Wk = (const float*)d_in[3];
    const float* bk = (const float*)d_in[4];
    const float* Wv = (const float*)d_in[5];
    const float* bv = (const float*)d_in[6];
    const float* Wo = (const float*)d_in[7];
    const float* bo = (const float*)d_in[8];
    float* out = (float*)d_out;

    const int E = E_DIM, S = S_LEN, B = B_SZ;
    const int M  = B * S;        // 8192 query rows
    const int Mk = B * SK_LEN;   // 2048 dilated rows

    // Workspace layout (bf16 intermediates), 256B aligned.
    size_t off = 0;
    char* ws = (char*)d_ws;
    auto alloc = [&](size_t bytes) -> void* {
        void* p = ws + off;
        off += (bytes + 255) & ~(size_t)255;
        return p;
    };
    bf16* xb   = (bf16*)alloc((size_t)M * E * sizeof(bf16));
    bf16* Wqt  = (bf16*)alloc((size_t)E * E * sizeof(bf16));
    bf16* Wkt  = (bf16*)alloc((size_t)E * E * sizeof(bf16));
    bf16* Wvt  = (bf16*)alloc((size_t)E * E * sizeof(bf16));
    bf16* Wot  = (bf16*)alloc((size_t)E * E * sizeof(bf16));
    bf16* Qb   = (bf16*)alloc((size_t)M * E * sizeof(bf16));
    bf16* Kdb  = (bf16*)alloc((size_t)Mk * E * sizeof(bf16));
    bf16* Vdb  = (bf16*)alloc((size_t)Mk * E * sizeof(bf16));
    bf16* Ctxb = (bf16*)alloc((size_t)M * E * sizeof(bf16));

    // 1) Precision staging.
    {
        const int n4 = (M * E) / 4;
        cvt_f32_to_bf16<<<(n4 + 255) / 256, 256, 0, stream>>>(x, xb, n4);
        dim3 tb(32, 8), tg(E / 32, E / 32);
        transpose_to_bf16<<<tg, tb, 0, stream>>>(Wq, Wqt, E);
        transpose_to_bf16<<<tg, tb, 0, stream>>>(Wk, Wkt, E);
        transpose_to_bf16<<<tg, tb, 0, stream>>>(Wv, Wvt, E);
        transpose_to_bf16<<<tg, tb, 0, stream>>>(Wo, Wot, E);
    }

    // 2) Projections. Dilation folded into K/V (project only rows s = 4*j).
    gemm_bf16_wmma<<<dim3(M / GE_TM,  E / GE_TN), 256, 0, stream>>>(
        xb, Wqt, bq, nullptr, Qb,  M,  E, E, S,      1);
    gemm_bf16_wmma<<<dim3(Mk / GE_TM, E / GE_TN), 256, 0, stream>>>(
        xb, Wkt, bk, nullptr, Kdb, Mk, E, E, SK_LEN, DIL);
    gemm_bf16_wmma<<<dim3(Mk / GE_TM, E / GE_TN), 256, 0, stream>>>(
        xb, Wvt, bv, nullptr, Vdb, Mk, E, E, SK_LEN, DIL);

    // 3) Attention: 32 query tiles x (B*H)=32 head slices.
    attn_dilated_wmma<<<dim3(S / 128, B * H_NUM), 256, 0, stream>>>(
        Qb, Kdb, Vdb, Ctxb);

    // 4) Output projection -> fp32 result.
    gemm_bf16_wmma<<<dim3(M / GE_TM, E / GE_TN), 256, 0, stream>>>(
        Ctxb, Wot, bo, out, nullptr, M, E, E, S, 1);
}
